// MultiHeadedAttention_4140348473883
// MI455X (gfx1250) — matrix operand order repaired and checked by reading, not compiled, not run on hardware
//
#include <hip/hip_runtime.h>
#include <hip/hip_bf16.h>

// ---------------- problem constants ----------------
#define B_   8
#define S_   2048
#define L_   2048
#define H_   4
#define DK   32
#define HID  128
#define ST   32     // s-tile per iteration
#define LT   32     // l rows per block (2 wave-halves x 16, x4 heads = 8 waves)
#define NT   (S_ / ST)

typedef __attribute__((ext_vector_type(16))) __bf16 v16bf;
typedef __attribute__((ext_vector_type(8)))  float  v8f;

union Frag32 {          // 32-byte WMMA A/B operand (16 x bf16)
    v16bf bf;
    uint4 q[2];
};

__device__ __forceinline__ unsigned short f2bf(float f) {
    unsigned u = __float_as_uint(f);
    u += 0x7fffu + ((u >> 16) & 1u);     // round-to-nearest-even
    return (unsigned short)(u >> 16);
}

// ---- CDNA5 async global->LDS (ISA §15.18.3 op 98, tracked by ASYNCcnt) ----
__device__ __forceinline__ void async_b128(const void* g, void* l) {
    unsigned lds            = (unsigned)(size_t)l;       // low 32b = LDS offset
    unsigned long long gadr = (unsigned long long)(size_t)g;
    asm volatile("global_load_async_to_lds_b128 %0, %1, off"
                 :: "v"(lds), "v"(gadr) : "memory");
}
// non-temporal variant for the read-once mask stream (ISA §7.3 TH=NT):
// don't let 134MB of mask evict the L2-resident h/v working set.
__device__ __forceinline__ void async_b128_nt(const void* g, void* l) {
    unsigned lds            = (unsigned)(size_t)l;
    unsigned long long gadr = (unsigned long long)(size_t)g;
    asm volatile("global_load_async_to_lds_b128 %0, %1, off th:TH_LOAD_NT"
                 :: "v"(lds), "v"(gadr) : "memory");
}
__device__ __forceinline__ void wait_async() {
    asm volatile("s_wait_asynccnt 0x0" ::: "memory");
}

// ---------------------------------------------------------------------------
// Stage 1: h = relu(k@W1 + q@W2 + bias) per head, bf16; v -> bf16 head-split.
// Layout: hbuf/vbuf[((b*H + head)*S + s)*DK + e]
// ---------------------------------------------------------------------------
__global__ __launch_bounds__(256) void prep_kernel(
    const float* __restrict__ q, const float* __restrict__ k,
    const float* __restrict__ v, const float* __restrict__ w1,
    const float* __restrict__ w2, const float* __restrict__ bias,
    unsigned short* __restrict__ hbuf, unsigned short* __restrict__ vbuf)
{
    __shared__ float W1s[32][32], W2s[32][32], bs[32];
    __shared__ float krow[2][HID], qrow[2][HID];
    const int tid = threadIdx.x;

    for (int i = tid; i < 1024; i += 256) {
        W1s[i >> 5][i & 31] = w1[i];
        W2s[i >> 5][i & 31] = w2[i];
    }
    if (tid < 32) bs[tid] = bias[tid];

    const size_t g0 = (size_t)blockIdx.x * 2;   // two (b,s) rows per block
    {
        int r = tid >> 7, c = tid & 127;
        krow[r][c] = k[(g0 + r) * HID + c];
        qrow[r][c] = q[(g0 + r) * HID + c];
    }
    __syncthreads();

    const int r  = tid >> 7;
    const int hd = (tid >> 5) & 3;
    const int e  = tid & 31;
    const size_t g = g0 + r;
    const int b = (int)(g / S_);
    const int s = (int)(g % S_);

    float acc = bs[e];
#pragma unroll
    for (int d = 0; d < 32; ++d)
        acc += krow[r][hd * 32 + d] * W1s[d][e] + qrow[r][hd * 32 + d] * W2s[d][e];
    acc = fmaxf(acc, 0.0f);

    const size_t o = (((size_t)b * H_ + hd) * S_ + s) * DK + e;
    hbuf[o] = f2bf(acc);
    vbuf[o] = f2bf(v[g * HID + hd * 32 + e]);
}

__global__ __launch_bounds__(256) void cvt_w0_kernel(
    const float* __restrict__ w0, unsigned short* __restrict__ w0b)
{
    int i = blockIdx.x * 256 + threadIdx.x;   // L_*DK = 65536 elements
    w0b[i] = f2bf(w0[i]);
}

// ---------------------------------------------------------------------------
// Stage 2: fused flash attention (max-free softmax, exp can't overflow here).
// Q = W0, K = h, V = v, softmax over s.  One block = (b, 32 l-rows) x 4 heads;
// mask tile streams once per block via async global->LDS, double-buffered.
// ---------------------------------------------------------------------------
__global__ __launch_bounds__(256) void attn_flash_kernel(
    const unsigned short* __restrict__ hbuf,
    const unsigned short* __restrict__ vbuf,
    const unsigned short* __restrict__ w0buf,
    const int* __restrict__ mask,
    float* __restrict__ out)
{
    // pads keep 16B LDS vectors aligned and spread banks
    __shared__ int            masktile[2][ST][36];        // [buf][s][l]
    __shared__ unsigned short htile[2][H_][ST][40];       // [buf][head][s][e]
    __shared__ unsigned short vT[2][H_][DK][40];          // [buf][head][d][s]
    __shared__ __bf16         pstage[8][16][40];          // [wave][l][s]

    const int tid   = threadIdx.x;
    const int wave  = tid >> 5;
    const int lane  = tid & 31;
    const int head  = wave >> 1;
    const int lhalf = wave & 1;
    const int lhi   = lane >> 4;
    const int llo   = lane & 15;

    const int b     = blockIdx.y;
    const int l0    = blockIdx.x * LT;
    const int lwave = l0 + lhalf * 16;

    const int* mbase = mask + ((size_t)b * S_) * L_ + l0;

    // ---- W0 A-fragment (loop-invariant): M=llo, K={k0..k0+15} ----
    Frag32 aw0;
    {
        const unsigned short* src = w0buf + (size_t)(lwave + llo) * DK;
        const int k0 = lhi * 16;
        aw0.q[0] = *(const uint4*)(src + k0);
        aw0.q[1] = *(const uint4*)(src + k0 + 8);
    }
    // ---- ones B-fragment for WMMA row-sums (every bf16 = 1.0) ----
    Frag32 bones;
    {
        uint4 o1s; o1s.x = o1s.y = o1s.z = o1s.w = 0x3f803f80u;
        bones.q[0] = o1s; bones.q[1] = o1s;
    }

    v8f o0 = {}, o1 = {}, dacc = {};

    // issue async mask/h loads + synchronous v transpose-scatter for tile t
    auto issue_tile = [&](int t) {
        const int s0 = t * ST, bi = t & 1;
        {
            int row = tid >> 3, seg = tid & 7;     // 32 rows x 8 x 16B
            async_b128_nt(mbase + (size_t)(s0 + row) * L_ + seg * 4,
                          &masktile[bi][row][seg * 4]);
        }
#pragma unroll
        for (int it = 0; it < 2; ++it) {           // 512 16B chunks of h and v
            int idx = tid + it * 256;
            int hh  = idx >> 7;
            int row = (idx >> 2) & 31;
            int seg = idx & 3;
            const size_t gsrc = (((size_t)b * H_ + hh) * S_ + s0 + row) * DK + seg * 8;
            async_b128(hbuf + gsrc, &htile[bi][hh][row][seg * 8]);
            uint4 vv = *(const uint4*)(vbuf + gsrc);
            const unsigned short* pv = (const unsigned short*)&vv;
#pragma unroll
            for (int k2 = 0; k2 < 8; ++k2)
                vT[bi][hh][seg * 8 + k2][row] = pv[k2];   // transpose into LDS
        }
    };

    issue_tile(0);

    for (int t = 0; t < NT; ++t) {
        const int bi = t & 1;
        wait_async();          // this wave's async stores for tile t complete
        __syncthreads();       // tile t visible to all; buffer bi^1 free
        if (t + 1 < NT) issue_tile(t + 1);

        // ---- scores: S[l,s] = W0 x h^T, two 16x16 tiles, K=32 per WMMA ----
        Frag32 bh0, bh1;
        {
            const int ecoff = lhi * 16;
            const unsigned short* hr0 = &htile[bi][head][llo][ecoff];
            bh0.q[0] = *(const uint4*)(hr0);
            bh0.q[1] = *(const uint4*)(hr0 + 8);
            const unsigned short* hr1 = &htile[bi][head][16 + llo][ecoff];
            bh1.q[0] = *(const uint4*)(hr1);
            bh1.q[1] = *(const uint4*)(hr1 + 8);
        }
        v8f z = {};
        v8f c0 = __builtin_amdgcn_wmma_f32_16x16x32_bf16(
            false, aw0.bf, false, bh0.bf, (short)0, z, false, false);
        v8f c1 = __builtin_amdgcn_wmma_f32_16x16x32_bf16(
            false, aw0.bf, false, bh1.bf, (short)0, z, false, false);

        // ---- masked exp (no max shift needed: |score| is O(1)) ----
#pragma unroll
        for (int j = 0; j < 8; ++j) {
            const int M    = j + 8 * lhi;
            const int lrow = lhalf * 16 + M;
            float p0 = (masktile[bi][llo][lrow]      != 0) ? __expf(c0[j]) : 0.0f;
            float p1 = (masktile[bi][16 + llo][lrow] != 0) ? __expf(c1[j]) : 0.0f;
            pstage[wave][M][llo]      = (__bf16)p0;   // C-layout -> A-layout via LDS
            pstage[wave][M][16 + llo] = (__bf16)p1;
        }

        // ---- A = P; rowsum += P x ones; O += P x V^T ----
        Frag32 ap, bv0, bv1;
        {
            const int k0 = lhi * 16;
            const __bf16* pr = &pstage[wave][llo][0];
            ap.q[0] = *(const uint4*)(pr + k0);
            ap.q[1] = *(const uint4*)(pr + k0 + 8);
            const int koff = lhi * 16;
            const unsigned short* v0r = &vT[bi][head][llo][koff];
            bv0.q[0] = *(const uint4*)(v0r);
            bv0.q[1] = *(const uint4*)(v0r + 8);
            const unsigned short* v1r = &vT[bi][head][16 + llo][koff];
            bv1.q[0] = *(const uint4*)(v1r);
            bv1.q[1] = *(const uint4*)(v1r + 8);
        }
        dacc = __builtin_amdgcn_wmma_f32_16x16x32_bf16(
            false, ap.bf, false, bones.bf, (short)0, dacc, false, false);
        o0 = __builtin_amdgcn_wmma_f32_16x16x32_bf16(
            false, ap.bf, false, bv0.bf, (short)0, o0, false, false);
        o1 = __builtin_amdgcn_wmma_f32_16x16x32_bf16(
            false, ap.bf, false, bv1.bf, (short)0, o1, false, false);
    }

    // ---- normalize and write out[b, l, head*32 + d] (write-once: NT stores) ----
#pragma unroll
    for (int j = 0; j < 8; ++j) {
        const int M  = j + 8 * lhi;
        const int lg = lwave + M;
        const float inv = 1.0f / dacc[j];     // rowsum lives in every column
        float* orow = out + ((size_t)b * L_ + lg) * HID + head * DK;
        __builtin_nontemporal_store(o0[j] * inv, orow + llo);
        __builtin_nontemporal_store(o1[j] * inv, orow + 16 + llo);
    }
}

// ---------------------------------------------------------------------------
extern "C" void kernel_launch(void* const* d_in, const int* in_sizes, int n_in,
                              void* d_out, int out_size, void* d_ws, size_t ws_size,
                              hipStream_t stream) {
    const float* q    = (const float*)d_in[0];
    const float* k    = (const float*)d_in[1];
    const float* v    = (const float*)d_in[2];
    const int*   mask = (const int*)  d_in[3];
    const float* w0   = (const float*)d_in[4];
    const float* w1   = (const float*)d_in[5];
    const float* w2   = (const float*)d_in[6];
    const float* bias = (const float*)d_in[7];
    float* out = (float*)d_out;

    // workspace: hbuf (4MB) | vbuf (4MB) | w0b (128KB)
    char* ws = (char*)d_ws;
    unsigned short* hbuf = (unsigned short*)ws;
    unsigned short* vbuf = (unsigned short*)(ws + (size_t)B_ * H_ * S_ * DK * 2);
    unsigned short* w0b  = (unsigned short*)(ws + (size_t)2 * B_ * H_ * S_ * DK * 2);

    prep_kernel<<<B_ * S_ / 2, 256, 0, stream>>>(q, k, v, w1, w2, bias, hbuf, vbuf);
    cvt_w0_kernel<<<(L_ * DK) / 256, 256, 0, stream>>>(w0, w0b);
    attn_flash_kernel<<<dim3(L_ / LT, B_), 256, 0, stream>>>(hbuf, vbuf, w0b, mask, out);
}
